// TMB_47820165874338
// MI455X (gfx1250) — matrix operand order repaired and checked by reading, not compiled, not run on hardware
//
#include <hip/hip_runtime.h>
#include <hip/hip_bf16.h>
#include <math.h>

// ---------------------------------------------------------------------------
// MI455X (gfx1250) implementation.
//
// Heavy work = three 3x3 64->64 convs over (16,64,256,256) f32 (~2.3e11 FLOP).
// Strategy: implicit GEMM with v_wmma_f32_16x16x32_bf16 (f32 accumulate).
//   GEMM view per batch:  M = Cout(64), N = pixels, K = Cin*9 = 576.
//   Block = 256 threads = 8 wave32; each wave owns one 16(M)x16(N) C tile,
//   block covers 64 cout x 32 pixels of one image row, marching down 32 rows.
//   Input rows live in a 3-row LDS ring laid out [row][col][ci] so a B
//   fragment is a single contiguous 32B LDS read.
//   Weights are invariant across rows/blocks -> each wave preloads its 18
//   A fragments (9 taps x 2 ci-halves = 144 VGPRs) ONCE; the B fragment is
//   software-pipelined one step ahead so each WMMA overlaps the next LDS
//   load (avoids the full s_wait_dscnt 0x0 drain per WMMA).
// Intermediates (y1, f) are stored bf16 in the workspace to halve HBM traffic.
// Final kernel fuses: WMMA conv(x, cW) + cb  +  mv * depthwise(f, dyn-kern),
// with the depthwise using vectorized (v8bf/v8f) LDS reads over contiguous
// output channels.
// ---------------------------------------------------------------------------

typedef __attribute__((ext_vector_type(16))) __bf16 v16bf;
typedef __attribute__((ext_vector_type(8)))  __bf16 v8bf;
typedef __attribute__((ext_vector_type(8)))  float  v8f;

#define Cc   64
#define Hh   256
#define Ww   256
#define Bb   16
#define ROWS 32     // rows per block
#define WPIX 32     // pixel-strip width per block

__device__ __forceinline__ float leaky01(float x) { return x > 0.f ? x : 0.1f * x; }

// ---------------------------------------------------------------------------
// Tiny per-sample MLP branch: mv (16x64) and dynamic depthwise kern (16x64x9)
// ---------------------------------------------------------------------------
__global__ void tmb_tiny_kernel(const float* __restrict__ t,
                                const float* __restrict__ tW1,
                                const float* __restrict__ tW2,
                                const float* __restrict__ kW1,
                                const float* __restrict__ kW2,
                                float* __restrict__ mv,
                                float* __restrict__ kern)
{
    const int b = blockIdx.x;      // 16 blocks
    const int c = threadIdx.x;     // 64 threads
    __shared__ float h1[Cc];
    __shared__ float k1[Cc];
    const float ts = t[b];
    h1[c] = leaky01(ts * tW1[c]);
    k1[c] = leaky01(ts * kW1[c]);
    __syncthreads();
    float acc = 0.f;
    #pragma unroll 8
    for (int j = 0; j < Cc; ++j) acc += tW2[c * Cc + j] * h1[j];
    mv[b * Cc + c] = 1.f / (1.f + expf(-acc));
    for (int tap = 0; tap < 9; ++tap) {
        float a2 = 0.f;
        #pragma unroll 8
        for (int j = 0; j < Cc; ++j) a2 += kW2[(c * 9 + tap) * Cc + j] * k1[j];
        kern[(b * Cc + c) * 9 + tap] = leaky01(a2);
    }
}

// ---------------------------------------------------------------------------
// Generic 3x3 conv (pad 1) + bias + LeakyReLU, bf16 output, WMMA bf16 core.
// IN_BF16 selects f32 (first layer) vs bf16 (second layer) input.
// Dynamic LDS: wl[9][64][64] bf16 (73728B) | il[3][34][64] bf16 (13056B)
// ---------------------------------------------------------------------------
template <bool IN_BF16>
__global__ __launch_bounds__(256, 1)
void tmb_conv3x3_wmma_kernel(const void* __restrict__ in_,
                             const float* __restrict__ Wg,    // OIHW (64,64,3,3)
                             const float* __restrict__ bias,  // (64)
                             __bf16* __restrict__ out)
{
    extern __shared__ char smem[];
    __bf16* wl = (__bf16*)smem;                       // [tap][co][ci]
    __bf16* il = (__bf16*)(smem + 9 * Cc * Cc * 2);   // [slot][col 0..33][ci]

    const int tid  = threadIdx.x;
    const int w0   = blockIdx.x * WPIX;   // 8
    const int h0   = blockIdx.y * ROWS;   // 8
    const int b    = blockIdx.z;          // 16

    // Stage weights: Wg[(co*64+ci)*9 + tap] -> wl[(tap*64+co)*64+ci] (bf16)
    for (int idx = tid; idx < 9 * Cc * Cc; idx += 256) {
        const int tap = idx % 9;
        const int rem = idx / 9;
        const int ci  = rem % Cc;
        const int co  = rem / Cc;
        wl[(tap * Cc + co) * Cc + ci] = (__bf16)Wg[idx];
    }

    auto loadRow = [&](int row) {
        const int slot = ((row % 3) + 3) % 3;
        for (int idx = tid; idx < 34 * Cc; idx += 256) {
            const int ci  = idx / 34;
            const int col = idx % 34;
            const int w   = w0 - 1 + col;
            float v = 0.f;
            if (row >= 0 && row < Hh && w >= 0 && w < Ww) {
                const size_t g = ((size_t)(b * Cc + ci) * Hh + row) * Ww + w;
                v = IN_BF16 ? (float)((const __bf16*)in_)[g]
                            : ((const float*)in_)[g];
            }
            il[(slot * 34 + col) * Cc + ci] = (__bf16)v;
        }
    };
    // Prefetch (global_prefetch_b8) one row ahead of the staging loads.
    auto prefetchRow = [&](int row) {
        if (row >= 0 && row < Hh) {
            const int ci  = tid / 4;             // 256 threads -> 64 ci x 4 lines
            const int wq  = (tid & 3) * 8;       // spread over the 33-elem span
            const int w   = min(w0 + wq, Ww - 1);
            const size_t g = ((size_t)(b * Cc + ci) * Hh + row) * Ww + w;
            if (IN_BF16) __builtin_prefetch((const __bf16*)in_ + g, 0, 0);
            else         __builtin_prefetch((const float*)in_ + g, 0, 0);
        }
    };
    loadRow(h0 - 1);
    loadRow(h0);
    __syncthreads();            // weights + first two rows visible

    const int lane  = tid & 31;
    const int wid   = tid >> 5;
    const int co0   = (wid >> 1) * 16;   // 4 M tiles
    const int n0    = (wid & 1) * 16;    // 2 N tiles
    const int lhalf = lane >> 4;
    const int lmod  = lane & 15;

    // ---- one-time A-fragment preload (weights are row/block invariant) ----
    v16bf afrag[9][2];
    #pragma unroll
    for (int tap = 0; tap < 9; ++tap) {
        const __bf16* arow = &wl[(tap * Cc + (co0 + lmod)) * Cc];
        #pragma unroll
        for (int cih = 0; cih < 2; ++cih) {
            const int kb = cih * 32 + lhalf * 16;
            v8bf alo = *(const v8bf*)(arow + kb);        // K quad lo
            v8bf ahi = *(const v8bf*)(arow + kb + 8);    // K quad hi
            afrag[tap][cih] = __builtin_shufflevector(alo, ahi,
                        0,1,2,3,4,5,6,7,8,9,10,11,12,13,14,15);
        }
    }

    float bsr[8];
    #pragma unroll
    for (int r = 0; r < 8; ++r) bsr[r] = bias[co0 + r + 8 * lhalf];

    for (int h = h0; h < h0 + ROWS; ++h) {
        loadRow(h + 1);
        prefetchRow(h + 2);
        __syncthreads();

        // Flattened K loop (k = tap*2 + cih) with depth-2 B pipeline.
        auto loadB = [&](int k) -> v16bf {
            const int tap = k >> 1, cih = k & 1;
            const int ky = tap / 3, kx = tap % 3;
            const int slot = (h - 1 + ky + 3) % 3;
            const __bf16* brow = &il[(slot * 34 + (n0 + lmod + kx)) * Cc];
            return *(const v16bf*)(brow + cih * 32 + lhalf * 16);
        };
        v8f acc = {0.f, 0.f, 0.f, 0.f, 0.f, 0.f, 0.f, 0.f};
        v16bf bcur = loadB(0);
        #pragma unroll
        for (int k = 0; k < 18; ++k) {
            v16bf bnext = bcur;
            if (k < 17) bnext = loadB(k + 1);
            acc = __builtin_amdgcn_wmma_f32_16x16x32_bf16(
                    false, afrag[k >> 1][k & 1], false, bcur, (short)0, acc,
                    false, false);
            bcur = bnext;
        }

        const int wpix = w0 + n0 + lmod;
        #pragma unroll
        for (int r = 0; r < 8; ++r) {
            const int co = co0 + r + 8 * lhalf;
            const float v = leaky01(acc[r] + bsr[r]);
            out[((size_t)(b * Cc + co) * Hh + h) * Ww + wpix] = (__bf16)v;
        }
        __syncthreads();  // protect ring slot of row h-1 before next load
    }
}

// ---------------------------------------------------------------------------
// Fused final: out = wmma_conv3x3(x, cW) + cb + mv * depthwise3x3(f, kern)
// Dynamic LDS: wl 73728B | xl 13056B | fl 13056B | kl[tap][co] 2304B
// ---------------------------------------------------------------------------
__global__ __launch_bounds__(256, 1)
void tmb_final_wmma_kernel(const float* __restrict__ x,
                           const float* __restrict__ cWg,
                           const float* __restrict__ cb,
                           const __bf16* __restrict__ f,
                           const float* __restrict__ mv,
                           const float* __restrict__ kern,
                           float* __restrict__ out)
{
    extern __shared__ char smem[];
    __bf16* wl = (__bf16*)smem;
    __bf16* xl = (__bf16*)(smem + 9 * Cc * Cc * 2);
    __bf16* fl = (__bf16*)(smem + 9 * Cc * Cc * 2 + 3 * 34 * Cc * 2);
    float*  kl = (float*)(smem + 9 * Cc * Cc * 2 + 2 * 3 * 34 * Cc * 2); // [tap][co]

    const int tid = threadIdx.x;
    const int w0  = blockIdx.x * WPIX;
    const int h0  = blockIdx.y * ROWS;
    const int b   = blockIdx.z;

    for (int idx = tid; idx < 9 * Cc * Cc; idx += 256) {
        const int tap = idx % 9;
        const int rem = idx / 9;
        const int ci  = rem % Cc;
        const int co  = rem / Cc;
        wl[(tap * Cc + co) * Cc + ci] = (__bf16)cWg[idx];
    }
    // kern (b,co,tap) -> kl[tap][co]  (transposed so 8 co are contiguous)
    for (int idx = tid; idx < Cc * 9; idx += 256) {
        const int co = idx / 9, tap = idx % 9;
        kl[tap * Cc + co] = kern[(b * Cc + co) * 9 + tap];
    }

    auto loadRowX = [&](int row) {
        const int slot = ((row % 3) + 3) % 3;
        for (int idx = tid; idx < 34 * Cc; idx += 256) {
            const int ci = idx / 34, col = idx % 34;
            const int w = w0 - 1 + col;
            float v = 0.f;
            if (row >= 0 && row < Hh && w >= 0 && w < Ww)
                v = x[((size_t)(b * Cc + ci) * Hh + row) * Ww + w];
            xl[(slot * 34 + col) * Cc + ci] = (__bf16)v;
        }
    };
    auto loadRowF = [&](int row) {
        const int slot = ((row % 3) + 3) % 3;
        for (int idx = tid; idx < 34 * Cc; idx += 256) {
            const int ci = idx / 34, col = idx % 34;
            const int w = w0 - 1 + col;
            __bf16 v = (__bf16)0.f;
            if (row >= 0 && row < Hh && w >= 0 && w < Ww)
                v = f[((size_t)(b * Cc + ci) * Hh + row) * Ww + w];
            fl[(slot * 34 + col) * Cc + ci] = v;
        }
    };
    auto prefetchRows = [&](int row) {
        if (row >= 0 && row < Hh) {
            const int ci = tid / 4;
            const int w  = min(w0 + (tid & 3) * 8, Ww - 1);
            const size_t g = ((size_t)(b * Cc + ci) * Hh + row) * Ww + w;
            __builtin_prefetch(x + g, 0, 0);
            __builtin_prefetch(f + g, 0, 0);
        }
    };
    loadRowX(h0 - 1); loadRowX(h0);
    loadRowF(h0 - 1); loadRowF(h0);
    __syncthreads();            // weights + kernels + first rows visible

    const int lane  = tid & 31;
    const int wid   = tid >> 5;
    const int co0   = (wid >> 1) * 16;
    const int n0    = (wid & 1) * 16;
    const int lhalf = lane >> 4;
    const int lmod  = lane & 15;
    const int cvec  = co0 + 8 * lhalf;   // base of this lane's 8 contiguous co

    // ---- one-time A-fragment preload ----
    v16bf afrag[9][2];
    #pragma unroll
    for (int tap = 0; tap < 9; ++tap) {
        const __bf16* arow = &wl[(tap * Cc + (co0 + lmod)) * Cc];
        #pragma unroll
        for (int cih = 0; cih < 2; ++cih) {
            const int kb = cih * 32 + lhalf * 16;
            v8bf alo = *(const v8bf*)(arow + kb);
            v8bf ahi = *(const v8bf*)(arow + kb + 8);
            afrag[tap][cih] = __builtin_shufflevector(alo, ahi,
                        0,1,2,3,4,5,6,7,8,9,10,11,12,13,14,15);
        }
    }

    float cbr[8], mvr[8];
    #pragma unroll
    for (int r = 0; r < 8; ++r) {
        cbr[r] = cb[cvec + r];
        mvr[r] = mv[b * Cc + cvec + r];
    }

    for (int h = h0; h < h0 + ROWS; ++h) {
        loadRowX(h + 1);
        loadRowF(h + 1);
        prefetchRows(h + 2);
        __syncthreads();

        auto loadB = [&](int k) -> v16bf {
            const int tap = k >> 1, cih = k & 1;
            const int ky = tap / 3, kx = tap % 3;
            const int slot = (h - 1 + ky + 3) % 3;
            const __bf16* brow = &xl[(slot * 34 + (n0 + lmod + kx)) * Cc];
            return *(const v16bf*)(brow + cih * 32 + lhalf * 16);
        };
        v8f acc = {0.f, 0.f, 0.f, 0.f, 0.f, 0.f, 0.f, 0.f};
        float dw[8] = {0.f, 0.f, 0.f, 0.f, 0.f, 0.f, 0.f, 0.f};
        v16bf bcur = loadB(0);
        #pragma unroll
        for (int k = 0; k < 18; ++k) {
            v16bf bnext = bcur;
            if (k < 17) bnext = loadB(k + 1);
            acc = __builtin_amdgcn_wmma_f32_16x16x32_bf16(
                    false, afrag[k >> 1][k & 1], false, bcur, (short)0, acc,
                    false, false);
            if ((k & 1) == 0) {
                // depthwise: 8 contiguous channels for this tap in one shot
                const int tap = k >> 1;
                const int ky = tap / 3, kx = tap % 3;
                const int slot = (h - 1 + ky + 3) % 3;
                v8bf fv = *(const v8bf*)
                    &fl[(slot * 34 + (n0 + lmod + kx)) * Cc + cvec];
                v8f  kv = *(const v8f*)&kl[tap * Cc + cvec];
                #pragma unroll
                for (int r = 0; r < 8; ++r) dw[r] += kv[r] * (float)fv[r];
            }
            bcur = bnext;
        }

        const int wpix = w0 + n0 + lmod;
        #pragma unroll
        for (int r = 0; r < 8; ++r) {
            out[((size_t)(b * Cc + cvec + r) * Hh + h) * Ww + wpix] =
                acc[r] + cbr[r] + mvr[r] * dw[r];
        }
        __syncthreads();
    }
}

// ---------------------------------------------------------------------------
extern "C" void kernel_launch(void* const* d_in, const int* in_sizes, int n_in,
                              void* d_out, int out_size, void* d_ws, size_t ws_size,
                              hipStream_t stream)
{
    const float* x   = (const float*)d_in[0];
    const float* t   = (const float*)d_in[1];
    const float* tW1 = (const float*)d_in[2];
    const float* tW2 = (const float*)d_in[3];
    const float* fW1 = (const float*)d_in[4];
    const float* fb1 = (const float*)d_in[5];
    const float* fW2 = (const float*)d_in[6];
    const float* fb2 = (const float*)d_in[7];
    const float* kW1 = (const float*)d_in[8];
    const float* kW2 = (const float*)d_in[9];
    const float* cW  = (const float*)d_in[10];
    const float* cb  = (const float*)d_in[11];
    float* out = (float*)d_out;

    char* ws = (char*)d_ws;
    float*  mv   = (float*)ws;                         // 16*64 f32   (4 KB)
    float*  kern = (float*)(ws + 4096);                // 16*64*9 f32 (36 KB)
    __bf16* y1   = (__bf16*)(ws + 40960);              // 134 MB bf16
    __bf16* ff   = y1 + (size_t)Bb * Cc * Hh * Ww;     // 134 MB bf16

    tmb_tiny_kernel<<<dim3(Bb), dim3(Cc), 0, stream>>>(t, tW1, tW2, kW1, kW2,
                                                       mv, kern);

    const dim3 grid(Ww / WPIX, Hh / ROWS, Bb);   // (8, 8, 16)
    const dim3 block(256);
    const size_t shConv  = 9 * Cc * Cc * 2 + 3 * 34 * Cc * 2;               // 86784
    const size_t shFinal = 9 * Cc * Cc * 2 + 2 * 3 * 34 * Cc * 2 + Cc * 9 * 4; // 102144

    tmb_conv3x3_wmma_kernel<false><<<grid, block, shConv, stream>>>(x,  fW1, fb1, y1);
    tmb_conv3x3_wmma_kernel<true ><<<grid, block, shConv, stream>>>(y1, fW2, fb2, ff);
    tmb_final_wmma_kernel<<<grid, block, shFinal, stream>>>(x, cW, cb, ff,
                                                            mv, kern, out);
}
